// MultiHeadAttentionConvIndex_10746008174998
// MI455X (gfx1250) — matrix operand order repaired and checked by reading, not compiled, not run on hardware
//
#include <hip/hip_runtime.h>
#include <hip/hip_bf16.h>

// ---------------------------------------------------------------------------
// MultiHeadAttentionConv for MI455X (gfx1250, wave32, WMMA bf16 16x16x32).
//
// Pipeline (all GEMMs on v_wmma_f32_16x16x32_bf16, fp32 accumulate):
//   prep      : weights -> transposed bf16 in ws; zero accumulators
//   edge_max  : per 16-edge tile: gather -> msg GEMM -> att GEMM -> atomicMax
//               of per-(dst,head) logit max (monotone-uint trick)
//   edge_acc  : recompute msg, alpha = exp(logit - segmax[dst]),
//               atomicAdd denom / count, vals GEMM, atomicAdd alpha*vals
//   node_out  : per 16-node tile: build [x | agg/denom | count | 0pad] (K=192)
//               GEMM with W_out^T (128x192) -> leaky_relu -> out
// ---------------------------------------------------------------------------

typedef __attribute__((ext_vector_type(16))) __bf16  bf16x16;
typedef __attribute__((ext_vector_type(8)))  float   f32x8;
typedef __attribute__((ext_vector_type(4)))  float   f32x4;
typedef __attribute__((ext_vector_type(4)))  unsigned u32x4;

union BFrag { bf16x16 v; u32x4 q[2]; };

__device__ inline unsigned short f2bf(float f) {            // RNE fp32->bf16
  unsigned u = __float_as_uint(f);
  u += 0x7FFFu + ((u >> 16) & 1u);
  return (unsigned short)(u >> 16);
}
__device__ inline float lrelu(float v) { return v >= 0.0f ? v : 0.01f * v; }

// monotone float<->uint map so fp32 max can use atomicMax(u32)
__device__ inline unsigned flipf(float f) {
  unsigned u = __float_as_uint(f);
  return (u & 0x80000000u) ? ~u : (u | 0x80000000u);
}
__device__ inline float unflipf(unsigned u) {
  return __uint_as_float((u & 0x80000000u) ? (u & 0x7FFFFFFFu) : ~u);
}

__device__ inline f32x8 wmma_bf16(bf16x16 a, bf16x16 b, f32x8 c) {
  return __builtin_amdgcn_wmma_f32_16x16x32_bf16(false, a, false, b,
                                                 (short)0, c, false, false);
}

// A fragment (16x32 bf16) from LDS row-major tile [16][stride].
// Lanes 0-15: row=lane, K = kbase+{0..15}; lanes 16-31: K = kbase+{16..31}.
__device__ inline bf16x16 load_a(const unsigned short* A, int stride, int lane, int kbase) {
  int m   = lane & 15;
  int klo = kbase + ((lane >> 4) << 4);
  BFrag f;
  f.q[0] = *(const u32x4*)(A + m * stride + klo);
  f.q[1] = *(const u32x4*)(A + m * stride + klo + 8);
  return f.v;
}

// B fragment (32x16 bf16) from global transposed weight W^T [Nrows][kstride].
// Lane holds column n = nbase+(lane&15); 16 consecutive K starting kbase(+16).
__device__ inline bf16x16 load_b(const unsigned short* __restrict__ W, int kstride,
                                 int lane, int nbase, int kbase) {
  int n = nbase + (lane & 15);
  int k = kbase + ((lane >> 4) << 4);
  const u32x4* p = (const u32x4*)(W + (size_t)n * kstride + k);
  BFrag f;
  f.q[0] = p[0];
  f.q[1] = p[1];
  return f.v;
}

#define AP  104   // LDS stride (halfwords) for 96-wide edge tiles
#define AP3 200   // LDS stride for 192-wide node tiles

// Gather [x[src]|edge_attr] for a 16-edge tile, run the pre-MLP GEMM, and
// leave msg = leaky(A@W_pre + b_pre) in LDS as a bf16 [16][AP] tile.
__device__ inline void build_msg(unsigned short* A,
                                 const float* __restrict__ x,
                                 const float* __restrict__ ea,
                                 const int* __restrict__ src,
                                 const unsigned short* __restrict__ WpreT,
                                 const float* __restrict__ b_pre,
                                 int tile, int E, int lane) {
  int erow = lane & 15, half = lane >> 4;
  int edge = tile * 16 + erow;
  if (edge >= E) edge = E - 1;

  { // gather x[src] -> cols [0,64), edge_attr -> cols [64,96)
    const f32x4* xr = (const f32x4*)(x + (size_t)src[edge] * 64);
#pragma unroll
    for (int j = 0; j < 8; ++j) {
      f32x4 v = xr[half * 8 + j];
      unsigned* p = (unsigned*)(A + erow * AP + half * 32 + j * 4);
      p[0] = (unsigned)f2bf(v.x) | ((unsigned)f2bf(v.y) << 16);
      p[1] = (unsigned)f2bf(v.z) | ((unsigned)f2bf(v.w) << 16);
    }
    const f32x4* er = (const f32x4*)(ea + (size_t)edge * 32);
#pragma unroll
    for (int j = 0; j < 4; ++j) {
      f32x4 v = er[half * 4 + j];
      unsigned* p = (unsigned*)(A + erow * AP + 64 + half * 16 + j * 4);
      p[0] = (unsigned)f2bf(v.x) | ((unsigned)f2bf(v.y) << 16);
      p[1] = (unsigned)f2bf(v.z) | ((unsigned)f2bf(v.w) << 16);
    }
  }
  __builtin_amdgcn_wave_barrier();

  const f32x8 zero = {0.f,0.f,0.f,0.f,0.f,0.f,0.f,0.f};
  f32x8 c[6] = {zero, zero, zero, zero, zero, zero};
#pragma unroll
  for (int ks = 0; ks < 3; ++ks) {
    bf16x16 a = load_a(A, AP, lane, ks * 32);
#pragma unroll
    for (int nt = 0; nt < 6; ++nt)
      c[nt] = wmma_bf16(a, load_b(WpreT, 96, lane, nt * 16, ks * 32), c[nt]);
  }
  __builtin_amdgcn_wave_barrier();

  // msg = leaky(c + b_pre) back into the same LDS tile (A fully consumed)
#pragma unroll
  for (int nt = 0; nt < 6; ++nt) {
    int n = nt * 16 + (lane & 15);
    float bb = b_pre[n];
#pragma unroll
    for (int r = 0; r < 8; ++r) {
      int m = r + (half << 3);
      A[m * AP + n] = f2bf(lrelu(c[nt][r] + bb));
    }
  }
  __builtin_amdgcn_wave_barrier();
}

// ---------------------------------------------------------------------------
__global__ void k_prep(const float* __restrict__ Wp, const float* __restrict__ Wa,
                       const float* __restrict__ Wv, const float* __restrict__ Wo,
                       unsigned short* WpreT, unsigned short* WattT,
                       unsigned short* WvalT, unsigned short* WoutT,
                       unsigned* segkey, float* denom, float* agg, float* cnt, int N) {
  long t  = (long)blockIdx.x * blockDim.x + threadIdx.x;
  long np = (long)gridDim.x * blockDim.x;
  for (long i = t; i < 96 * 96; i += np) {
    int n = (int)(i / 96), k = (int)(i % 96);
    WpreT[i] = f2bf(Wp[k * 96 + n]);
    WvalT[i] = f2bf(Wv[k * 96 + n]);
  }
  for (long i = t; i < 16 * 96; i += np) {
    int n = (int)(i / 96), k = (int)(i % 96);
    WattT[i] = (n < 4) ? f2bf(Wa[k * 4 + n]) : (unsigned short)0;
  }
  for (long i = t; i < 128 * 192; i += np) {
    int n = (int)(i / 192), k = (int)(i % 192);
    WoutT[i] = (k < 161) ? f2bf(Wo[k * 128 + n]) : (unsigned short)0;
  }
  for (long i = t; i < (long)N * 4; i += np) { segkey[i] = 0u; denom[i] = 0.0f; }
  for (long i = t; i < (long)N; i += np)      cnt[i] = 0.0f;
  for (long i = t; i < (long)N * 96; i += np) agg[i] = 0.0f;
}

// ---------------------------------------------------------------------------
__global__ __launch_bounds__(256) void k_edge_max(
    const float* __restrict__ x, const float* __restrict__ ea,
    const int* __restrict__ eidx,
    const unsigned short* __restrict__ WpreT, const unsigned short* __restrict__ WattT,
    const float* __restrict__ b_pre, const float* __restrict__ b_att,
    unsigned* __restrict__ segkey, int E) {
  __shared__ __align__(16) unsigned short sA[8][16 * AP];
  int wave = threadIdx.x >> 5, lane = threadIdx.x & 31;
  int tile = blockIdx.x * 8 + wave;
  if (tile * 16 >= E) return;
  const int* src = eidx;
  const int* dst = eidx + E;
  unsigned short* A = sA[wave];
  int half = lane >> 4;

  build_msg(A, x, ea, src, WpreT, b_pre, tile, E, lane);

  const f32x8 zero = {0.f,0.f,0.f,0.f,0.f,0.f,0.f,0.f};
  f32x8 cl = zero;
#pragma unroll
  for (int ks = 0; ks < 3; ++ks) {
    bf16x16 a = load_a(A, AP, lane, ks * 32);
    cl = wmma_bf16(a, load_b(WattT, 96, lane, 0, ks * 32), cl);
  }
  int h = lane & 15;
  if (h < 4) {
    float ba = b_att[h];
#pragma unroll
    for (int r = 0; r < 8; ++r) {
      int m = r + (half << 3);
      int e2 = tile * 16 + m;
      if (e2 < E) {
        float lg = cl[r] + ba;
        atomicMax(segkey + (size_t)dst[e2] * 4 + h, flipf(lg));
      }
    }
  }
}

// ---------------------------------------------------------------------------
__global__ __launch_bounds__(256) void k_edge_acc(
    const float* __restrict__ x, const float* __restrict__ ea,
    const int* __restrict__ eidx,
    const unsigned short* __restrict__ WpreT, const unsigned short* __restrict__ WattT,
    const unsigned short* __restrict__ WvalT,
    const float* __restrict__ b_pre, const float* __restrict__ b_att,
    const float* __restrict__ b_val,
    const unsigned* __restrict__ segkey,
    float* __restrict__ denom, float* __restrict__ agg, float* __restrict__ cnt, int E) {
  __shared__ __align__(16) unsigned short sA[8][16 * AP];
  __shared__ float sAl[8][64];
  __shared__ int   sD[8][16];
  int wave = threadIdx.x >> 5, lane = threadIdx.x & 31;
  int tile = blockIdx.x * 8 + wave;
  if (tile * 16 >= E) return;
  const int* src = eidx;
  const int* dst = eidx + E;
  unsigned short* A = sA[wave];
  int half = lane >> 4;

  build_msg(A, x, ea, src, WpreT, b_pre, tile, E, lane);

  if (lane < 16) {
    int e = tile * 16 + lane;
    sD[wave][lane] = dst[e < E ? e : E - 1];
  }

  const f32x8 zero = {0.f,0.f,0.f,0.f,0.f,0.f,0.f,0.f};
  f32x8 cl = zero;
#pragma unroll
  for (int ks = 0; ks < 3; ++ks) {
    bf16x16 a = load_a(A, AP, lane, ks * 32);
    cl = wmma_bf16(a, load_b(WattT, 96, lane, 0, ks * 32), cl);
  }
  __builtin_amdgcn_wave_barrier();

  int h = lane & 15;
  if (h < 4) {
    float ba = b_att[h];
#pragma unroll
    for (int r = 0; r < 8; ++r) {
      int m = r + (half << 3);
      int e2 = tile * 16 + m;
      int d  = sD[wave][m];
      float lg = cl[r] + ba;
      float a  = __expf(lg - unflipf(segkey[(size_t)d * 4 + h]));
      if (e2 < E) {
        sAl[wave][m * 4 + h] = a;
        atomicAdd(denom + (size_t)d * 4 + h, a);
      } else {
        sAl[wave][m * 4 + h] = 0.0f;
      }
    }
  }
  if (lane < 16 && tile * 16 + lane < E) atomicAdd(cnt + sD[wave][lane], 1.0f);
  __builtin_amdgcn_wave_barrier();

  // vals GEMM, scale by alpha, scatter-accumulate into agg[dst]
  f32x8 cv[6] = {zero, zero, zero, zero, zero, zero};
#pragma unroll
  for (int ks = 0; ks < 3; ++ks) {
    bf16x16 a = load_a(A, AP, lane, ks * 32);
#pragma unroll
    for (int nt = 0; nt < 6; ++nt)
      cv[nt] = wmma_bf16(a, load_b(WvalT, 96, lane, nt * 16, ks * 32), cv[nt]);
  }
#pragma unroll
  for (int nt = 0; nt < 6; ++nt) {
    int n  = nt * 16 + (lane & 15);
    float bb = b_val[n];
    int hh = n / 24;
#pragma unroll
    for (int r = 0; r < 8; ++r) {
      int m  = r + (half << 3);
      int e2 = tile * 16 + m;
      if (e2 < E) {
        float v = cv[nt][r] + bb;
        float a = sAl[wave][m * 4 + hh];
        atomicAdd(agg + (size_t)sD[wave][m] * 96 + n, a * v);
      }
    }
  }
}

// ---------------------------------------------------------------------------
__global__ __launch_bounds__(256) void k_node_out(
    const float* __restrict__ x, const float* __restrict__ agg,
    const float* __restrict__ denom, const float* __restrict__ cnt,
    const unsigned short* __restrict__ WoutT, const float* __restrict__ b_out,
    float* __restrict__ out, int N) {
  __shared__ __align__(16) unsigned short sA[8][16 * AP3];
  int wave = threadIdx.x >> 5, lane = threadIdx.x & 31;
  int tile = blockIdx.x * 8 + wave;
  if (tile * 16 >= N) return;
  int m = lane & 15, half = lane >> 4;
  int node = tile * 16 + m;
  unsigned short* A = sA[wave];

  if (node < N) {
    float inv[4];
#pragma unroll
    for (int hh = 0; hh < 4; ++hh) {
      float d = denom[(size_t)node * 4 + hh];
      inv[hh] = (d != 0.0f) ? 1.0f / d : 0.0f;
    }
    if (half == 0) {
      const f32x4* xr = (const f32x4*)(x + (size_t)node * 64);
#pragma unroll
      for (int j = 0; j < 16; ++j) {
        f32x4 v = xr[j];
        unsigned* p = (unsigned*)(A + m * AP3 + j * 4);
        p[0] = (unsigned)f2bf(v.x) | ((unsigned)f2bf(v.y) << 16);
        p[1] = (unsigned)f2bf(v.z) | ((unsigned)f2bf(v.w) << 16);
      }
      const f32x4* ar = (const f32x4*)(agg + (size_t)node * 96);
#pragma unroll
      for (int j = 0; j < 8; ++j) {
        f32x4 v = ar[j];
        int c0 = j * 4;
        float a0 = v.x * inv[(c0 + 0) / 24], a1 = v.y * inv[(c0 + 1) / 24];
        float a2 = v.z * inv[(c0 + 2) / 24], a3 = v.w * inv[(c0 + 3) / 24];
        unsigned* p = (unsigned*)(A + m * AP3 + 64 + c0);
        p[0] = (unsigned)f2bf(a0) | ((unsigned)f2bf(a1) << 16);
        p[1] = (unsigned)f2bf(a2) | ((unsigned)f2bf(a3) << 16);
      }
    } else {
      const f32x4* ar = (const f32x4*)(agg + (size_t)node * 96);
#pragma unroll
      for (int j = 8; j < 24; ++j) {
        f32x4 v = ar[j];
        int c0 = j * 4;
        float a0 = v.x * inv[(c0 + 0) / 24], a1 = v.y * inv[(c0 + 1) / 24];
        float a2 = v.z * inv[(c0 + 2) / 24], a3 = v.w * inv[(c0 + 3) / 24];
        unsigned* p = (unsigned*)(A + m * AP3 + 64 + c0);
        p[0] = (unsigned)f2bf(a0) | ((unsigned)f2bf(a1) << 16);
        p[1] = (unsigned)f2bf(a2) | ((unsigned)f2bf(a3) << 16);
      }
      A[m * AP3 + 160] = f2bf(cnt[node]);
#pragma unroll
      for (int c = 161; c < 192; ++c) A[m * AP3 + c] = 0;
    }
  } else {
    for (int c = half * 96; c < half * 96 + 96; ++c) A[m * AP3 + c] = 0;
  }
  __builtin_amdgcn_wave_barrier();

  const f32x8 zero = {0.f,0.f,0.f,0.f,0.f,0.f,0.f,0.f};
  f32x8 c[8] = {zero, zero, zero, zero, zero, zero, zero, zero};
#pragma unroll
  for (int ks = 0; ks < 6; ++ks) {
    bf16x16 a = load_a(A, AP3, lane, ks * 32);
#pragma unroll
    for (int nt = 0; nt < 8; ++nt)
      c[nt] = wmma_bf16(a, load_b(WoutT, 192, lane, nt * 16, ks * 32), c[nt]);
  }
#pragma unroll
  for (int nt = 0; nt < 8; ++nt) {
    int n = nt * 16 + (lane & 15);
    float bb = b_out[n];
#pragma unroll
    for (int r = 0; r < 8; ++r) {
      int mm = r + (half << 3);
      int nd = tile * 16 + mm;
      if (nd < N) out[(size_t)nd * 128 + n] = lrelu(c[nt][r] + bb);
    }
  }
}

// ---------------------------------------------------------------------------
extern "C" void kernel_launch(void* const* d_in, const int* in_sizes, int n_in,
                              void* d_out, int out_size, void* d_ws, size_t ws_size,
                              hipStream_t stream) {
  const float* x     = (const float*)d_in[0];
  const float* ea    = (const float*)d_in[1];
  const int*   eidx  = (const int*)d_in[2];
  const float* W_pre = (const float*)d_in[3];
  const float* b_pre = (const float*)d_in[4];
  const float* W_att = (const float*)d_in[5];
  const float* b_att = (const float*)d_in[6];
  const float* W_val = (const float*)d_in[7];
  const float* b_val = (const float*)d_in[8];
  const float* W_out = (const float*)d_in[9];
  const float* b_out = (const float*)d_in[10];
  float* out = (float*)d_out;

  const int N = in_sizes[0] / 64;
  const int E = in_sizes[1] / 32;
  (void)n_in; (void)out_size; (void)ws_size;

  char* ws = (char*)d_ws;
  size_t off = 0;
  auto take = [&](size_t bytes) -> char* {
    char* p = ws + off;
    off = (off + bytes + 255) & ~(size_t)255;
    return p;
  };
  unsigned short* WpreT = (unsigned short*)take(96 * 96 * 2);
  unsigned short* WattT = (unsigned short*)take(16 * 96 * 2);
  unsigned short* WvalT = (unsigned short*)take(96 * 96 * 2);
  unsigned short* WoutT = (unsigned short*)take(128 * 192 * 2);
  unsigned* segkey = (unsigned*)take((size_t)N * 4 * 4);
  float* denom = (float*)take((size_t)N * 4 * 4);
  float* cnt   = (float*)take((size_t)N * 4);
  float* agg   = (float*)take((size_t)N * 96 * 4);   // total ws ~42.5 MB

  k_prep<<<1024, 256, 0, stream>>>(W_pre, W_att, W_val, W_out,
                                   WpreT, WattT, WvalT, WoutT,
                                   segkey, denom, agg, cnt, N);
  int eb = (E + 127) / 128;
  k_edge_max<<<eb, 256, 0, stream>>>(x, ea, eidx, WpreT, WattT, b_pre, b_att,
                                     segkey, E);
  k_edge_acc<<<eb, 256, 0, stream>>>(x, ea, eidx, WpreT, WattT, WvalT,
                                     b_pre, b_att, b_val, segkey,
                                     denom, agg, cnt, E);
  int nb = (((N + 15) / 16) + 7) / 8;
  k_node_out<<<nb, 256, 0, stream>>>(x, agg, denom, cnt, WoutT, b_out, out, N);
}
